// Net_1632087572623
// MI455X (gfx1250) — compile-verified, matrix operand order checked by reading, not run on hardware
//
#include <hip/hip_runtime.h>
#include <hip/hip_bf16.h>

typedef __attribute__((ext_vector_type(16))) _Float16 v16h;
typedef __attribute__((ext_vector_type(8)))  float    v8f;

#define BB  8
#define NN  2048
#define KNN 20
#define NUM_CLASSES 40

// ---------------------------------------------------------------------------
// A-fragment loader: 16-bit A 16x32 layout (lanes 0-15: M=lane, K-halves per
// ISA table). aT is a row-major [16 x ld] f16 tile in LDS.
__device__ inline v16h load_a_frag(const _Float16* aT, int ld, int kc, int lane) {
    int m  = lane & 15;
    int kh = lane >> 4;
    const _Float16* p0 = aT + m * ld + kc * 32 + kh * 8;
    union { v16h v; _Float16 h[16]; } u;
#pragma unroll
    for (int j = 0; j < 8; ++j) { u.h[j] = p0[j]; u.h[8 + j] = p0[16 + j]; }
    return u.v;
}

// ---------------------------------------------------------------------------
// One dense layer on a 16-row tile: D = relu(A*W + b) -> LDS (f16).
template<int KDIM, int H, int LD>
__device__ inline void wave_gemm_relu(const _Float16* aT,
                                      const _Float16* __restrict__ wp,
                                      const float* __restrict__ bias,
                                      _Float16* oT, int lane) {
    constexpr int KC = KDIM / 32, NT = H / 16;
    int nn = lane & 15, half = lane >> 4;
    for (int nt = 0; nt < NT; ++nt) {
        v8f acc = {};
#pragma unroll
        for (int kc = 0; kc < KC; ++kc) {
            v16h a = load_a_frag(aT, LD, kc, lane);
            v16h b = *(const v16h*)(wp + (((size_t)kc * NT + nt) * 32 + lane) * 16);
            acc = __builtin_amdgcn_wmma_f32_16x16x32_f16(
                false, a, false, b, (short)0, acc, false, false);
        }
        float bv = bias[nt * 16 + nn];
#pragma unroll
        for (int r = 0; r < 8; ++r) {
            float v = fmaxf(acc[r] + bv, 0.f);                  // v_max_num_f32
            oT[(r + 8 * half) * LD + nt * 16 + nn] = (_Float16)v;
        }
    }
}

// Final layer of a tile: relu then atomic-max aggregation into out[row/DIV][*].
// ReLU outputs are >= 0 so uint-bitpattern max == float max, zero init valid.
// DIV is a compile-time constant so the row->dest division strength-reduces;
// dest[] is hoisted out of the nt loop (independent of nt).
template<int KDIM, int H, int LD, int DIV>
__device__ inline void wave_gemm_atomicmax(const _Float16* aT,
                                           const _Float16* __restrict__ wp,
                                           const float* __restrict__ bias,
                                           float* __restrict__ out,
                                           long rowBase, int lane) {
    constexpr int KC = KDIM / 32, NT = H / 16;
    int nn = lane & 15, half = lane >> 4;
    long destRow[8];
#pragma unroll
    for (int r = 0; r < 8; ++r)
        destRow[r] = ((rowBase + r + 8 * half) / DIV) * (long)H;
    for (int nt = 0; nt < NT; ++nt) {
        v8f acc = {};
#pragma unroll
        for (int kc = 0; kc < KC; ++kc) {
            v16h a = load_a_frag(aT, LD, kc, lane);
            v16h b = *(const v16h*)(wp + (((size_t)kc * NT + nt) * 32 + lane) * 16);
            acc = __builtin_amdgcn_wmma_f32_16x16x32_f16(
                false, a, false, b, (short)0, acc, false, false);
        }
        float bv = bias[nt * 16 + nn];
#pragma unroll
        for (int r = 0; r < 8; ++r) {
            float v = fmaxf(acc[r] + bv, 0.f);
            atomicMax((unsigned int*)(out + destRow[r] + nt * 16 + nn),
                      __float_as_uint(v));
        }
    }
}

// ---------------------------------------------------------------------------
// Repack fp32 weight [Krows x H] into f16 B-fragment order, K padded to KP.
__global__ void pack_w(const float* __restrict__ w, int Krows, int KP, int H,
                       _Float16* __restrict__ out) {
    int tid = blockIdx.x * blockDim.x + threadIdx.x;
    if (tid >= KP * H) return;
    int j    = tid & 15;
    int l    = (tid >> 4) & 31;
    int rest = tid >> 9;
    int NT   = H >> 4;
    int nt   = rest % NT;
    int kc   = rest / NT;
    int n    = nt * 16 + (l & 15);
    int kh   = l >> 4;
    int k    = kc * 32 + ((j < 8) ? (kh * 8 + j) : (16 + kh * 8 + (j - 8)));
    float v  = (k < Krows) ? w[k * H + n] : 0.0f;
    out[tid] = (_Float16)v;
}

// ---------------------------------------------------------------------------
// kNN: thread-per-row scan over LDS-staged 128-point chunks; top-20 insertion.
template<int D>
__global__ __launch_bounds__(128) void knn_kernel(const float* __restrict__ x,
                                                  int* __restrict__ idxo) {
    constexpr int ROWS = 128, CH = 128;
    __shared__ float chunk[CH * D];
    __shared__ float bd[ROWS][KNN];
    __shared__ int   bi[ROWS][KNN];
    int tid = threadIdx.x;
    int blocksPerCloud = NN / ROWS;
    int b        = blockIdx.x / blocksPerCloud;
    int rowStart = (blockIdx.x % blocksPerCloud) * ROWS;
    size_t cloudBase = (size_t)b * NN;
    int n = rowStart + tid;

    float xi[D];
#pragma unroll
    for (int f = 0; f < D; ++f) xi[f] = x[(cloudBase + n) * D + f];
    for (int kk = 0; kk < KNN; ++kk) { bd[tid][kk] = 3.4e38f; bi[tid][kk] = 0; }

    for (int c = 0; c < NN / CH; ++c) {
        __syncthreads();
        for (int t = tid; t < CH * D; t += 128)
            chunk[t] = x[(cloudBase + (size_t)c * CH) * D + t];
        if (c + 1 < NN / CH)
            __builtin_prefetch(&x[(cloudBase + (size_t)(c + 1) * CH) * D + tid], 0, 0);
        __syncthreads();
        for (int j = 0; j < CH; ++j) {
            float d2 = 0.f;
#pragma unroll
            for (int f = 0; f < D; ++f) {
                float df = xi[f] - chunk[j * D + f];
                d2 = fmaf(df, df, d2);
            }
            if (d2 < bd[tid][KNN - 1]) {
                int p = KNN - 1;
                while (p > 0 && bd[tid][p - 1] > d2) {
                    bd[tid][p] = bd[tid][p - 1];
                    bi[tid][p] = bi[tid][p - 1];
                    --p;
                }
                bd[tid][p] = d2;
                bi[tid][p] = c * CH + j;   // cloud-local neighbor index
            }
        }
    }
    for (int kk = 0; kk < KNN; ++kk)
        idxo[(cloudBase + n) * KNN + kk] = bi[tid][kk];
}

// ---------------------------------------------------------------------------
// DynamicEdgeConv: build [xi, xj-xi] edge tiles, 3-layer WMMA MLP, max-agg
// fused via atomic max. One wave per 16-edge tile, 4 waves per block.
template<int DIN, int DIN2, int H1, int H2, int H3, int LD>
__global__ __launch_bounds__(128) void edgeconv_kernel(
        const float* __restrict__ x, const int* __restrict__ idx,
        const _Float16* __restrict__ w1p, const float* __restrict__ b1,
        const _Float16* __restrict__ w2p, const float* __restrict__ b2,
        const _Float16* __restrict__ w3p, const float* __restrict__ b3,
        float* __restrict__ out) {
    __shared__ _Float16 smem[4][2][16 * LD];
    int lane = threadIdx.x & 31;
    int wv   = threadIdx.x >> 5;
    long tile     = (long)blockIdx.x * 4 + wv;
    long edgeBase = tile * 16;
    _Float16* A    = smem[wv][0];
    _Float16* Hbuf = smem[wv][1];

    // Build edge-feature tile [16 x DIN2] (zero-padded to DIN2)
    for (int r = 0; r < 16; ++r) {
        long e  = edgeBase + r;
        long pt = e / KNN;
        int  k  = (int)(e % KNN);
        long b  = pt / NN;
        int  nb = idx[pt * KNN + k];
        long nr = b * NN + nb;
        for (int f = lane; f < DIN2; f += 32) {
            float v;
            if (f < DIN)            v = x[pt * DIN + f];
            else if (f < 2 * DIN)   v = x[nr * DIN + (f - DIN)] - x[pt * DIN + (f - DIN)];
            else                    v = 0.f;
            A[r * LD + f] = (_Float16)v;
        }
    }
    __syncthreads();
    wave_gemm_relu<DIN2, H1, LD>(A, w1p, b1, Hbuf, lane);
    __syncthreads();
    wave_gemm_relu<H1, H2, LD>(Hbuf, w2p, b2, A, lane);
    __syncthreads();
    wave_gemm_atomicmax<H2, H3, LD, KNN>(A, w3p, b3, out, edgeBase, lane);
}

// ---------------------------------------------------------------------------
// Head l0: [B*N,256] @ [256,512] + relu, fused global max-pool -> pooled[B,512]
__global__ __launch_bounds__(128) void head_l0_kernel(
        const float* __restrict__ x2, const _Float16* __restrict__ wp,
        const float* __restrict__ bias, float* __restrict__ pooled) {
    __shared__ _Float16 sA[4][16 * 256];
    int lane = threadIdx.x & 31;
    int wv   = threadIdx.x >> 5;
    long tile    = (long)blockIdx.x * 4 + wv;
    long rowBase = tile * 16;
    _Float16* A = sA[wv];
    for (int r = 0; r < 16; ++r) {
        long row = rowBase + r;
        for (int f = lane; f < 256; f += 32)
            A[r * 256 + f] = (_Float16)x2[row * 256 + f];
    }
    __syncthreads();
    wave_gemm_atomicmax<256, 512, 256, NN>(A, wp, bias, pooled, rowBase, lane);
}

// ---------------------------------------------------------------------------
// Tiny 8-row classifier head + log_softmax (fp32 VALU).
__global__ __launch_bounds__(256) void head_final_kernel(
        const float* __restrict__ pooled,
        const float* __restrict__ l1w, const float* __restrict__ l1b,
        const float* __restrict__ l2w, const float* __restrict__ l2b,
        const float* __restrict__ l3w, const float* __restrict__ l3b,
        float* __restrict__ outp) {
    __shared__ float sp[8 * 512];
    __shared__ float h1[8 * 256];
    __shared__ float h2[8 * 256];
    __shared__ float lg[8 * NUM_CLASSES];
    int tid = threadIdx.x;
    for (int t = tid; t < 8 * 512; t += 256) sp[t] = pooled[t];
    __syncthreads();
    for (int o = tid; o < 8 * 256; o += 256) {
        int r = o >> 8, c = o & 255;
        float a = l1b[c];
        for (int k = 0; k < 512; ++k) a = fmaf(sp[r * 512 + k], l1w[k * 256 + c], a);
        h1[o] = fmaxf(a, 0.f);
    }
    __syncthreads();
    for (int o = tid; o < 8 * 256; o += 256) {
        int r = o >> 8, c = o & 255;
        float a = l2b[c];
        for (int k = 0; k < 256; ++k) a = fmaf(h1[r * 256 + k], l2w[k * 256 + c], a);
        h2[o] = fmaxf(a, 0.f);
    }
    __syncthreads();
    for (int o = tid; o < 8 * NUM_CLASSES; o += 256) {
        int r = o / NUM_CLASSES, c = o % NUM_CLASSES;
        float a = l3b[c];
        for (int k = 0; k < 256; ++k) a = fmaf(h2[r * 256 + k], l3w[k * NUM_CLASSES + c], a);
        lg[o] = a;
    }
    __syncthreads();
    if (tid < 8) {
        float m = -3.4e38f;
        for (int c = 0; c < NUM_CLASSES; ++c) m = fmaxf(m, lg[tid * NUM_CLASSES + c]);
        float s = 0.f;
        for (int c = 0; c < NUM_CLASSES; ++c) s += expf(lg[tid * NUM_CLASSES + c] - m);
        float ls = logf(s);
        for (int c = 0; c < NUM_CLASSES; ++c)
            outp[tid * NUM_CLASSES + c] = lg[tid * NUM_CLASSES + c] - m - ls;
    }
}

// ---------------------------------------------------------------------------
extern "C" void kernel_launch(void* const* d_in, const int* in_sizes, int n_in,
                              void* d_out, int out_size, void* d_ws, size_t ws_size,
                              hipStream_t stream) {
    (void)in_sizes; (void)n_in; (void)out_size; (void)ws_size;
    const float* pos  = (const float*)d_in[0];
    // d_in[1] = batch (int64), contiguous equal partition -> ignored
    const float* c1w1 = (const float*)d_in[2];  const float* c1b1 = (const float*)d_in[3];
    const float* c1w2 = (const float*)d_in[4];  const float* c1b2 = (const float*)d_in[5];
    const float* c1w3 = (const float*)d_in[6];  const float* c1b3 = (const float*)d_in[7];
    const float* c2w1 = (const float*)d_in[8];  const float* c2b1 = (const float*)d_in[9];
    const float* c2w2 = (const float*)d_in[10]; const float* c2b2 = (const float*)d_in[11];
    const float* c2w3 = (const float*)d_in[12]; const float* c2b3 = (const float*)d_in[13];
    const float* l0w  = (const float*)d_in[14]; const float* l0b  = (const float*)d_in[15];
    const float* l1w  = (const float*)d_in[16]; const float* l1b  = (const float*)d_in[17];
    const float* l2w  = (const float*)d_in[18]; const float* l2b  = (const float*)d_in[19];
    const float* l3w  = (const float*)d_in[20]; const float* l3b  = (const float*)d_in[21];

    char* ws = (char*)d_ws;
    // workspace layout (bytes)
    const size_t OFF_X1   = 0;                            // 16384*64*4  = 4,194,304
    const size_t OFF_X2   = 4194304;                      // 16384*256*4 = 16,777,216
    const size_t OFF_POOL = 20971520;                     // 8*512*4     = 16,384
    const size_t OFF_I1   = 20987904;                     // 16384*20*4  = 1,310,720
    const size_t OFF_I2   = 22298624;                     // 1,310,720
    const size_t OFF_WP   = 23609344;                     // packed f16 weights

    float* x1     = (float*)(ws + OFF_X1);
    float* x2     = (float*)(ws + OFF_X2);
    float* pooled = (float*)(ws + OFF_POOL);
    int*   idx1   = (int*)(ws + OFF_I1);
    int*   idx2   = (int*)(ws + OFF_I2);
    _Float16* c1w1p = (_Float16*)(ws + OFF_WP + 0);        // 32x64   -> 4096 B
    _Float16* c1w2p = (_Float16*)(ws + OFF_WP + 4096);     // 64x64   -> 8192 B
    _Float16* c1w3p = (_Float16*)(ws + OFF_WP + 12288);    // 64x64   -> 8192 B
    _Float16* c2w1p = (_Float16*)(ws + OFF_WP + 20480);    // 128x128 -> 32768 B
    _Float16* c2w2p = (_Float16*)(ws + OFF_WP + 53248);    // 128x128 -> 32768 B
    _Float16* c2w3p = (_Float16*)(ws + OFF_WP + 86016);    // 128x256 -> 65536 B
    _Float16* l0p   = (_Float16*)(ws + OFF_WP + 151552);   // 256x512 -> 262144 B

    // zero-init atomic-max accumulators (relu outputs >= 0, 0-bits == 0.0f)
    hipMemsetAsync(x1, 0, (size_t)BB * NN * 64 * 4, stream);
    hipMemsetAsync(x2, 0, (size_t)BB * NN * 256 * 4, stream);
    hipMemsetAsync(pooled, 0, (size_t)BB * 512 * 4, stream);

    // repack weights to f16 B-fragment order
    pack_w<<<8,   256, 0, stream>>>(c1w1,   6,  32,  64, c1w1p);
    pack_w<<<16,  256, 0, stream>>>(c1w2,  64,  64,  64, c1w2p);
    pack_w<<<16,  256, 0, stream>>>(c1w3,  64,  64,  64, c1w3p);
    pack_w<<<64,  256, 0, stream>>>(c2w1, 128, 128, 128, c2w1p);
    pack_w<<<64,  256, 0, stream>>>(c2w2, 128, 128, 128, c2w2p);
    pack_w<<<128, 256, 0, stream>>>(c2w3, 128, 128, 256, c2w3p);
    pack_w<<<512, 256, 0, stream>>>(l0w,  256, 256, 512, l0p);

    // conv1: kNN on pos (D=3) then edge MLP 6->64->64->64
    knn_kernel<3><<<BB * (NN / 128), 128, 0, stream>>>(pos, idx1);
    edgeconv_kernel<3, 32, 64, 64, 64, 64><<<5120, 128, 0, stream>>>(
        pos, idx1, c1w1p, c1b1, c1w2p, c1b2, c1w3p, c1b3, x1);

    // conv2: kNN on features (D=64) then edge MLP 128->128->128->256
    knn_kernel<64><<<BB * (NN / 128), 128, 0, stream>>>(x1, idx2);
    edgeconv_kernel<64, 128, 128, 128, 256, 128><<<5120, 128, 0, stream>>>(
        x1, idx2, c2w1p, c2b1, c2w2p, c2b2, c2w3p, c2b3, x2);

    // head l0 + global max pool, then tiny classifier + log_softmax
    head_l0_kernel<<<256, 128, 0, stream>>>(x2, l0p, l0b, pooled);
    head_final_kernel<<<1, 256, 0, stream>>>(pooled, l1w, l1b, l2w, l2b,
                                             l3w, l3b, (float*)d_out);
}
